// BasePNARetriever_8555574853794
// MI455X (gfx1250) — compile-verified, matrix operand order checked by reading, not run on hardware
//
#include <hip/hip_runtime.h>

#define VOCAB 32000
#define HID   4096
#define RDIM  64
#define NTOK  50000
#define SLEN  16
#define KRE   768            // 12 * RDIM
#define KC    256            // K-chunk staged into LDS per double-buffer slot
#define KHALF (HID / 2)      // split-K: each grid-Y half covers 2048
#define NCH   (KHALF / KC)   // 8 chunks per half

typedef __attribute__((ext_vector_type(16))) __bf16 v16bf;
typedef __attribute__((ext_vector_type(8)))  __bf16 v8bf;
typedef __attribute__((ext_vector_type(8)))  float  v8f;
typedef __attribute__((ext_vector_type(4)))  float  v4f;

// ---------------------------------------------------------------------------
// Kernel 0: pre-convert W_down and W_re to bf16 tables in workspace.
// ---------------------------------------------------------------------------
__global__ __launch_bounds__(256) void k0_cvt(const float* __restrict__ Wd,
                                              const float* __restrict__ Wre,
                                              __bf16* __restrict__ wdb,
                                              __bf16* __restrict__ wreb) {
    const int i = blockIdx.x * 256 + threadIdx.x;
    if (i < RDIM * HID) wdb[i]  = (__bf16)Wd[i];
    if (i < RDIM * KRE) wreb[i] = (__bf16)Wre[i];
}

// ---------------------------------------------------------------------------
// Kernel 1: partial[y][32000,64] = E[:, y*2048:(y+1)*2048] @ W_down_half^T
// Split-K over gridDim.y (2 halves -> 4000 waves for latency hiding).
// One wave -> one 16(M) x 64(N) tile. B (bf16) staged to LDS with
// async-to-LDS, double buffered; A is the raw f32 HBM stream.
// ---------------------------------------------------------------------------
__global__ __launch_bounds__(256) void k1_down_gemm(const float* __restrict__ E,
                                                    const __bf16* __restrict__ Wdb,
                                                    float* __restrict__ partial) {
    extern __shared__ __bf16 smem[];                 // 2 buffers * 64 * 256 bf16 = 64 KB
    const int lane  = threadIdx.x & 31;
    const int wave  = threadIdx.x >> 5;
    const int mb    = (blockIdx.x * 8 + wave) * 16;  // 250 blocks * 128 rows = 32000
    const int mrow  = mb + (lane & 15);
    const int khlf  = (lane >> 4) * 8;               // 0 (lanes 0-15) / 8 (lanes 16-31)
    const int kbase = blockIdx.y * KHALF;
    const int kend  = kbase + KHALF;

    const float* __restrict__ erow = E + (size_t)mrow * HID;
    float* __restrict__ pout = partial + (size_t)blockIdx.y * VOCAB * RDIM;
    const unsigned long long wd_base = (unsigned long long)(const void*)Wdb;

    // Stage one 64x256 bf16 chunk of W_down into LDS buffer `buf` (32 KB):
    // 2048 x 16-byte lane transfers over 256 threads x 8 iterations.
    auto stage = [&](int chunk, int buf) {
        const unsigned kc       = (unsigned)(kbase + chunk * KC);
        const unsigned lds_base = (unsigned)buf * (64u * KC * 2u);
#pragma unroll
        for (int t = 0; t < 8; ++t) {
            unsigned u     = (unsigned)t * 256u + threadIdx.x;     // 0..2047
            unsigned n     = u >> 5;                               // B row (N)
            unsigned ku    = u & 31;                               // 8-elem unit in K
            unsigned g_off = (n * HID + kc + ku * 8u) * 2u;        // bytes
            unsigned l_off = lds_base + u * 16u;                   // bytes
            asm volatile("global_load_async_to_lds_b128 %0, %1, %2"
                         :
                         : "v"(l_off), "v"(g_off), "s"(wd_base)
                         : "memory");
        }
    };

    v8f acc[4] = {v8f{}, v8f{}, v8f{}, v8f{}};

    stage(0, 0);
    for (int c = 0; c < NCH; ++c) {
        asm volatile("s_wait_asynccnt 0x0" ::: "memory");  // chunk c transfers done
        __syncthreads();                                   // visible to whole block
        if (c + 1 < NCH) stage(c + 1, (c + 1) & 1);        // overlap next chunk

        const __bf16* __restrict__ bs = smem + (c & 1) * (64 * KC);
        const int kcg = kbase + c * KC;
        for (int kl = 0; kl < KC; kl += 32) {
            if (kcg + kl + 2 * KC < kend)                  // A-stream prefetch (into L2)
                __builtin_prefetch(erow + kcg + kl + 2 * KC, 0, 0);

            // ---- issue A global loads
            const v4f* pa0 = (const v4f*)(erow + kcg + kl + khlf);
            const v4f* pa1 = (const v4f*)(erow + kcg + kl + 16 + khlf);
            v4f x0 = pa0[0], x1 = pa0[1], y0 = pa1[0], y1 = pa1[1];

            // ---- issue ALL B ds loads up front (one dscnt wait, not four)
            v8bf lo[4], hi[4];
#pragma unroll
            for (int nt = 0; nt < 4; ++nt) {
                const v8bf* p = (const v8bf*)(bs + (nt * 16 + (lane & 15)) * KC + kl + khlf);
                lo[nt] = p[0];
                hi[nt] = p[2];                             // +16 elements
            }

            // ---- convert A while loads are in flight
            v16bf a;
#pragma unroll
            for (int e = 0; e < 4; ++e) {
                a[e]      = (__bf16)x0[e];
                a[4 + e]  = (__bf16)x1[e];
                a[8 + e]  = (__bf16)y0[e];
                a[12 + e] = (__bf16)y1[e];
            }

            // ---- 4 back-to-back WMMAs
#pragma unroll
            for (int nt = 0; nt < 4; ++nt) {
                v16bf b;
#pragma unroll
                for (int e = 0; e < 8; ++e) { b[e] = lo[nt][e]; b[8 + e] = hi[nt][e]; }
                acc[nt] = __builtin_amdgcn_wmma_f32_16x16x32_bf16(
                    false, a, false, b, (short)0, acc[nt], false, false);
            }
        }
        __syncthreads();   // everyone done reading this buffer before re-stage
    }

    // ---- store 16x64 partial tile (C layout: elem i -> row i (+8 upper lanes))
    const int nlo = lane & 15;
    const int mhi = (lane >> 4) * 8;
#pragma unroll
    for (int i = 0; i < 8; ++i) {
        float* orow = pout + (size_t)(mb + mhi + i) * RDIM + nlo;
        orow[0]  = acc[0][i];
        orow[16] = acc[1][i];
        orow[32] = acc[2][i];
        orow[48] = acc[3][i];
    }
}

// ---------------------------------------------------------------------------
// Kernel 1b: rtab = partial0 + partial1 (vectorized, in place on partial0)
// ---------------------------------------------------------------------------
__global__ __launch_bounds__(256) void k1b_add(float* __restrict__ p0,
                                               const float* __restrict__ p1) {
    const int i = blockIdx.x * 256 + threadIdx.x;           // float4 index
    if (i < (VOCAB * RDIM) / 4) {
        v4f a = ((const v4f*)p0)[i];
        v4f b = ((const v4f*)p1)[i];
        ((v4f*)p0)[i] = a + b;
    }
}

// ---------------------------------------------------------------------------
// Kernel 2: gather + masked pooling.  thread -> (batch, r) ; 4 batches/block.
// feat[b] = [mean(64) | max(64) | min(64) | std(64)],  logs[b] = log(len)
// ---------------------------------------------------------------------------
__global__ __launch_bounds__(256) void k2_pool(const float* __restrict__ Rtab,
                                               const int* __restrict__ tokens,
                                               float* __restrict__ feat,
                                               float* __restrict__ logs) {
    const int b = blockIdx.x * 4 + (threadIdx.x >> 6);
    const int r = threadIdx.x & 63;
    if (b >= NTOK) return;

    const int* idx = tokens + (size_t)b * SLEN;
    float sum = 0.f, sq = 0.f, mx = -1e10f, mn = 1e10f;
    int len = 0;
#pragma unroll
    for (int s = 0; s < SLEN; ++s) {
        int tk = idx[s];
        if (tk > 0) {                       // wave-uniform branch
            float v = Rtab[(size_t)tk * RDIM + r];
            sum += v;
            sq  += v * v;
            mx   = fmaxf(mx, v);
            mn   = fminf(mn, v);
            ++len;
        }
    }
    float fl   = (float)len;
    float inv  = 1.0f / (fl + 1e-10f);
    float mean = sum * inv;
    float var  = sq * inv - mean * mean;
    float sd   = sqrtf(fmaxf(var, 1e-6f));

    float* fr = feat + (size_t)b * 256;
    fr[r]       = mean;
    fr[64 + r]  = mx;
    fr[128 + r] = mn;
    fr[192 + r] = sd;
    if (r == 0) logs[b] = logf(fl);
}

// ---------------------------------------------------------------------------
// Kernel 2b: deterministic mean of logs[] (single block, LDS tree reduce)
// ---------------------------------------------------------------------------
__global__ __launch_bounds__(256) void k2b_mean(const float* __restrict__ logs,
                                                float* __restrict__ meanp) {
    __shared__ float red[256];
    float s = 0.f;
    for (int i = threadIdx.x; i < NTOK; i += 256) s += logs[i];
    red[threadIdx.x] = s;
    __syncthreads();
#pragma unroll
    for (int off = 128; off > 0; off >>= 1) {
        if ((int)threadIdx.x < off) red[threadIdx.x] += red[threadIdx.x + off];
        __syncthreads();
    }
    if (threadIdx.x == 0) meanp[0] = red[0] / (float)NTOK;
}

// ---------------------------------------------------------------------------
// Kernel 3: out[50000,64] = (feat x scales)[50000,768] @ W_re[64,768]^T + b_re
// scales fold into the A operand on the fly: A(m, k) = feat[m][k/3]*sc[k%3].
// B comes from the pre-converted bf16 W_re (L2-resident).
// ---------------------------------------------------------------------------
__global__ __launch_bounds__(256) void k3_regemm(const float* __restrict__ feat,
                                                 const float* __restrict__ logs,
                                                 const float* __restrict__ meanp,
                                                 const __bf16* __restrict__ wreb,
                                                 const float* __restrict__ bre,
                                                 float* __restrict__ out) {
    const int lane = threadIdx.x & 31;
    const int wave = threadIdx.x >> 5;
    const int bb   = (blockIdx.x * 8 + wave) * 16;      // batch tile base
    if (bb >= NTOK) return;                              // whole-wave uniform exit

    const int m    = lane & 15;
    const int b    = bb + m;
    const int khlf = (lane >> 4) * 8;

    const float ml  = meanp[0];
    const float sc1 = logs[b] / (ml + 1e-10f);
    const float sc[3] = {1.0f, sc1, 1.0f / fmaxf(sc1, 0.01f)};

    const float* __restrict__ frow = feat + (size_t)b * 256;
    const __bf16* __restrict__ wrowb[4];
#pragma unroll
    for (int nt = 0; nt < 4; ++nt)
        wrowb[nt] = wreb + (size_t)(nt * 16 + m) * KRE;

    v8f acc[4] = {v8f{}, v8f{}, v8f{}, v8f{}};

    for (int kb = 0; kb < KRE; kb += 32) {
        // ---- issue all B loads first
        v8bf lo[4], hi[4];
#pragma unroll
        for (int nt = 0; nt < 4; ++nt) {
            const v8bf* p = (const v8bf*)(wrowb[nt] + kb + khlf);
            lo[nt] = p[0];
            hi[nt] = p[2];
        }
        // ---- build A (scattered f32 reads, L0-resident row) while B in flight
        v16bf a;
#pragma unroll
        for (int e = 0; e < 8; ++e) {
            int k0 = kb + khlf + e;
            int k1 = kb + 16 + khlf + e;
            a[e]     = (__bf16)(frow[k0 / 3] * sc[k0 % 3]);
            a[8 + e] = (__bf16)(frow[k1 / 3] * sc[k1 % 3]);
        }
#pragma unroll
        for (int nt = 0; nt < 4; ++nt) {
            v16bf bm;
#pragma unroll
            for (int e = 0; e < 8; ++e) { bm[e] = lo[nt][e]; bm[8 + e] = hi[nt][e]; }
            acc[nt] = __builtin_amdgcn_wmma_f32_16x16x32_bf16(
                false, a, false, bm, (short)0, acc[nt], false, false);
        }
    }

    const int nlo = lane & 15;
    const int mhi = (lane >> 4) * 8;
#pragma unroll
    for (int nt = 0; nt < 4; ++nt) {
        float bias = bre[nt * 16 + nlo];
#pragma unroll
        for (int i = 0; i < 8; ++i)
            out[(size_t)(bb + mhi + i) * RDIM + nt * 16 + nlo] = acc[nt][i] + bias;
    }
}

// ---------------------------------------------------------------------------
// Kernel 4: row L2-normalize in place. One wave per 64-wide row.
// ---------------------------------------------------------------------------
__global__ __launch_bounds__(256) void k4_norm(float* __restrict__ out) {
    const int lane = threadIdx.x & 31;
    const int wave = threadIdx.x >> 5;
    const int row  = blockIdx.x * 8 + wave;
    if (row >= NTOK) return;

    float2* rp = (float2*)(out + (size_t)row * RDIM);
    float2 v = rp[lane];
    float s = v.x * v.x + v.y * v.y;
#pragma unroll
    for (int off = 16; off > 0; off >>= 1)
        s += __shfl_xor(s, off, 32);
    float inv = 1.0f / fmaxf(sqrtf(s), 1e-12f);
    v.x *= inv;
    v.y *= inv;
    rp[lane] = v;
}

// ---------------------------------------------------------------------------
// Launch
// ---------------------------------------------------------------------------
extern "C" void kernel_launch(void* const* d_in, const int* in_sizes, int n_in,
                              void* d_out, int out_size, void* d_ws, size_t ws_size,
                              hipStream_t stream) {
    const float* E    = (const float*)d_in[0];   // [VOCAB, HID]
    const int*   tok  = (const int*)  d_in[1];   // [NTOK, SLEN]
    const float* Wd   = (const float*)d_in[2];   // [RDIM, HID]
    const float* Wre  = (const float*)d_in[3];   // [RDIM, 12*RDIM]
    const float* bre  = (const float*)d_in[4];   // [RDIM]
    float*       out  = (float*)d_out;           // [NTOK, RDIM]

    float*  rtab  = (float*)d_ws;                          // partial0 / final Rtab
    float*  p1    = rtab + (size_t)VOCAB * RDIM;           // partial1
    float*  feat  = p1 + (size_t)VOCAB * RDIM;             // 50000*256 f32
    __bf16* wdb   = (__bf16*)(feat + (size_t)NTOK * 256);  // 64*4096 bf16
    __bf16* wreb  = wdb + (size_t)RDIM * HID;              // 64*768 bf16
    float*  logs  = (float*)(wreb + (size_t)RDIM * KRE);   // 50000 f32
    float*  meanp = logs + NTOK;                           // 1 f32

    k0_cvt<<<(RDIM * HID + 255) / 256, 256, 0, stream>>>(Wd, Wre, wdb, wreb);
    k1_down_gemm<<<dim3(VOCAB / 128, 2), 256, 2 * 64 * KC * sizeof(__bf16), stream>>>(E, wdb, rtab);
    k1b_add<<<(VOCAB * RDIM / 4 + 255) / 256, 256, 0, stream>>>(rtab, p1);
    k2_pool<<<(NTOK + 3) / 4, 256, 0, stream>>>(rtab, tok, feat, logs);
    k2b_mean<<<1, 256, 0, stream>>>(logs, meanp);
    k3_regemm<<<(NTOK + 127) / 128, 256, 0, stream>>>(feat, logs, meanp, wreb, bre, out);
    k4_norm<<<(NTOK + 7) / 8, 256, 0, stream>>>(out);
}
